// ClassifierGenerator_53386443489627
// MI455X (gfx1250) — matrix operand order repaired and checked by reading, not compiled, not run on hardware
//
#include <hip/hip_runtime.h>
#include <hip/hip_bf16.h>

// ---------------------------------------------------------------------------
// Types for CDNA5 WMMA (wave32): v_wmma_f32_16x16x32_bf16
// ---------------------------------------------------------------------------
typedef __attribute__((ext_vector_type(16))) __bf16 v16bf;
typedef __attribute__((ext_vector_type(8)))  __bf16 v8bf;
typedef __attribute__((ext_vector_type(8)))  float  v8f;

union V16 {            // one 16x32 (A) or 32x16 (B) bf16 fragment
    v16bf v;
    v8bf  h[2];
    unsigned short u[16];
};
union V8 {
    v8bf v;
    unsigned short u[8];
};

// native fp32 -> bf16 (lowers to v_cvt_(pk_)bf16_f32 on gfx1250, RNE)
__device__ __forceinline__ unsigned short bfbits(float f) {
    __bf16 h = (__bf16)f;
    return __builtin_bit_cast(unsigned short, h);
}
// pack two fp32 -> one dword of two bf16 (lo, hi)
__device__ __forceinline__ unsigned pack_bf2(float lo, float hi) {
    return (unsigned)bfbits(lo) | ((unsigned)bfbits(hi) << 16);
}

__device__ __forceinline__ v8f wmma_bf16(v16bf a, v16bf b, v8f c) {
    return __builtin_amdgcn_wmma_f32_16x16x32_bf16(
        /*neg_a=*/false, a, /*neg_b=*/false, b,
        /*c_mod=*/(short)0, c, /*reuse_a=*/false, /*reuse_b=*/false);
}

// load 8 consecutive fp32 (two b128 loads) -> bf16 half-fragment, K-guarded
__device__ __forceinline__ v8bf load8_bf(const float* __restrict__ rowp, int k, int kmax) {
    V8 r;
    if (k + 7 < kmax) {
        float4 a = *reinterpret_cast<const float4*>(rowp + k);
        float4 b = *reinterpret_cast<const float4*>(rowp + k + 4);
        r.u[0] = bfbits(a.x); r.u[1] = bfbits(a.y); r.u[2] = bfbits(a.z); r.u[3] = bfbits(a.w);
        r.u[4] = bfbits(b.x); r.u[5] = bfbits(b.y); r.u[6] = bfbits(b.z); r.u[7] = bfbits(b.w);
    } else {
#pragma unroll
        for (int i = 0; i < 8; ++i) {
            float f = (k + i < kmax) ? rowp[k + i] : 0.0f;
            r.u[i] = bfbits(f);
        }
    }
    return r.v;
}

#define LSEQ 1024   // L for every tensor in this net

// ---------------------------------------------------------------------------
// Fused 1x1 conv:  Y[b,co,l] = act( sum_ci W[co,ci]*X[b,ci,l] + bias[co] (+R) )
// Tile: M=32 (Cout) x N=64 (L), 4 waves / 128 threads, K in steps of 32.
// act: 0=none, 1=relu, 2=10*relu
// ---------------------------------------------------------------------------
__global__ __launch_bounds__(128)
void conv1x1_wmma(const float* __restrict__ X, const float* __restrict__ W,
                  const float* __restrict__ bias, const float* __restrict__ resid,
                  float* __restrict__ Y, int Cin, int Cout, int act)
{
    const int b     = blockIdx.z;
    const int mBase = blockIdx.y * 32;
    const int l0    = blockIdx.x * 64;
    const int tid   = threadIdx.x;
    const int wave  = tid >> 5;
    const int lane  = tid & 31;
    const int lhalf = lane & 15;
    const int hi    = lane >> 4;

    const float* Xb = X + (size_t)b * Cin * LSEQ;
    float*       Yb = Y + (size_t)b * Cout * LSEQ;
    const float* Rb = resid ? resid + (size_t)b * Cout * LSEQ : nullptr;

    // X tile transposed to [l][k] bf16, pitch 48 halves (96B, 16B aligned)
    __shared__ __align__(16) unsigned short tX[64 * 48];

    const int mSub = mBase + (wave >> 1) * 16;     // this wave's 16 output rows
    const int nSub = (wave & 1) * 32;              // this wave's 32 cols (2 frags)
    const int kA0  = hi * 8;                       // A-frag K interleave base
    const int kB0  = hi * 8;                       // B-frag K interleave base

    v8f acc0 = {}; v8f acc1 = {};
    const bool rowsValid = (mSub < Cout);

    for (int k0 = 0; k0 < Cin; k0 += 32) {
        // ---- cooperative stage: X[k0..k0+31][l0..l0+63] -> tX[l][k] (bf16) ----
        // 256 pair-slots: each covers k-rows (2k,2k+1) x 4 columns; b32 stores
#pragma unroll
        for (int it = 0; it < 2; ++it) {
            int idx = it * 128 + tid;         // 0..255
            int k   = (idx >> 4) * 2;         // 0,2,..,30
            int l4  = (idx & 15) * 4;         // 0..60
            float4 a, c;
            int kk = k0 + k;
            if (kk < Cin) {
                a = *reinterpret_cast<const float4*>(&Xb[(size_t)kk * LSEQ + l0 + l4]);
                if (k0 + 32 < Cin)            // CDNA5 prefetch of next K slab
                    __builtin_prefetch(&Xb[(size_t)(kk + 32) * LSEQ + l0 + l4], 0, 3);
            } else a = make_float4(0.f, 0.f, 0.f, 0.f);
            if (kk + 1 < Cin)
                c = *reinterpret_cast<const float4*>(&Xb[(size_t)(kk + 1) * LSEQ + l0 + l4]);
            else c = make_float4(0.f, 0.f, 0.f, 0.f);
            unsigned* dst0 = reinterpret_cast<unsigned*>(&tX[(l4 + 0) * 48 + k]);
            unsigned* dst1 = reinterpret_cast<unsigned*>(&tX[(l4 + 1) * 48 + k]);
            unsigned* dst2 = reinterpret_cast<unsigned*>(&tX[(l4 + 2) * 48 + k]);
            unsigned* dst3 = reinterpret_cast<unsigned*>(&tX[(l4 + 3) * 48 + k]);
            *dst0 = pack_bf2(a.x, c.x);
            *dst1 = pack_bf2(a.y, c.y);
            *dst2 = pack_bf2(a.z, c.z);
            *dst3 = pack_bf2(a.w, c.w);
        }
        __syncthreads();

        // ---- A fragment from weights (rows K-contiguous in memory) ----
        V16 A;
        if (rowsValid) {
            const float* wrow = W + (size_t)(mSub + lhalf) * Cin;
            A.h[0] = load8_bf(wrow, k0 + kA0,      Cin);
            A.h[1] = load8_bf(wrow, k0 + 16 + kA0, Cin);
        } else {
#pragma unroll
            for (int i = 0; i < 16; ++i) A.u[i] = 0;
        }

        // ---- B fragments from LDS (transposed tile) ----
        V16 B0, B1;
        {
            const unsigned short* p0 = &tX[(nSub      + lhalf) * 48 + kB0];
            const unsigned short* p1 = &tX[(nSub + 16 + lhalf) * 48 + kB0];
            B0.h[0] = *reinterpret_cast<const v8bf*>(p0);
            B0.h[1] = *reinterpret_cast<const v8bf*>(p0 + 16);
            B1.h[0] = *reinterpret_cast<const v8bf*>(p1);
            B1.h[1] = *reinterpret_cast<const v8bf*>(p1 + 16);
        }

        acc0 = wmma_bf16(A.v, B0.v, acc0);
        acc1 = wmma_bf16(A.v, B1.v, acc1);
        __syncthreads();
    }

    // ---- epilogue: bias, residual, activation, store ----
    const int col0 = l0 + nSub + lhalf;
    const int col1 = col0 + 16;
#pragma unroll
    for (int vv = 0; vv < 8; ++vv) {
        int row = mSub + hi * 8 + vv;          // C-layout row for this VGPR
        if (row < Cout) {
            float bv = bias[row];
            float o0 = acc0[vv] + bv;
            float o1 = acc1[vv] + bv;
            if (Rb) {
                o0 += Rb[(size_t)row * LSEQ + col0];
                o1 += Rb[(size_t)row * LSEQ + col1];
            }
            if (act >= 1) { o0 = fmaxf(o0, 0.f); o1 = fmaxf(o1, 0.f); }
            if (act == 2) { o0 *= 10.f; o1 *= 10.f; }
            Yb[(size_t)row * LSEQ + col0] = o0;
            Yb[(size_t)row * LSEQ + col1] = o1;
        }
    }
}

// ---------------------------------------------------------------------------
// Flash-style attention head:
//   z[f,q] = sum_k fk[k,f] qk[k,q] / sqrt(32); w = exp(clip(z,±30))
//   out[v,q] = (sum_f fv[v,f] w[f,q]) / (sum_f w[f,q] + 1e-16)
// Block: 128 threads / 4 waves; wave owns 16 queries; f loop in steps of 32.
// ---------------------------------------------------------------------------
__global__ __launch_bounds__(128)
void attn_head_wmma(const float* __restrict__ FK, const float* __restrict__ FV,
                    const float* __restrict__ QK, float* __restrict__ Z, int rowOff)
{
    const int NK = 32, NV = 128;
    const int b    = blockIdx.y;
    const int tid  = threadIdx.x;
    const int wave = tid >> 5;
    const int lane = tid & 31;
    const int lhalf = lane & 15;
    const int hi    = lane >> 4;
    const int q0    = blockIdx.x * 64 + wave * 16;

    const float* fkb = FK + (size_t)b * NK * LSEQ;
    const float* fvb = FV + (size_t)b * NV * LSEQ;
    const float* qkb = QK + (size_t)b * NK * LSEQ;
    float*       zb  = Z  + (size_t)b * 512 * LSEQ + (size_t)rowOff * LSEQ;

    __shared__ __align__(16) unsigned short tFK[32 * 48];      // [f][k] shared by all waves
    __shared__ __align__(16) unsigned short tW[4][16 * 48];    // per-wave [q][f] exp weights

    const int kA0 = hi * 8;
    const int kB0 = hi * 8;

    // qk B-fragment: fixed over the f loop. col q = q0+lhalf, K = kB0..kB0+7, 16+kB0..16+kB0+7
    V16 Bq;
    {
        const float* qp = qkb + (q0 + lhalf);
#pragma unroll
        for (int i = 0; i < 16; ++i)
            Bq.u[i] = bfbits(qp[(size_t)(kB0 + i + (i >> 3) * 8) * LSEQ]);
    }

    v8f acc[8] = {{}, {}, {}, {}, {}, {}, {}, {}};
    float den = 0.0f;

    for (int f0 = 0; f0 < LSEQ; f0 += 32) {
        // ---- stage fk[0..31][f0..f0+31] -> tFK[f][k] (bf16 transposed) ----
        // 128 pair-slots: k-rows (2k,2k+1) x 4 field columns; b32 stores
        {
            int k  = (tid >> 3) * 2;        // 0,2,..,30
            int f4 = (tid & 7) * 4;         // 0..28
            float4 a = *reinterpret_cast<const float4*>(&fkb[(size_t)k * LSEQ + f0 + f4]);
            float4 c = *reinterpret_cast<const float4*>(&fkb[(size_t)(k + 1) * LSEQ + f0 + f4]);
            *reinterpret_cast<unsigned*>(&tFK[(f4 + 0) * 48 + k]) = pack_bf2(a.x, c.x);
            *reinterpret_cast<unsigned*>(&tFK[(f4 + 1) * 48 + k]) = pack_bf2(a.y, c.y);
            *reinterpret_cast<unsigned*>(&tFK[(f4 + 2) * 48 + k]) = pack_bf2(a.z, c.z);
            *reinterpret_cast<unsigned*>(&tFK[(f4 + 3) * 48 + k]) = pack_bf2(a.w, c.w);
        }
        __syncthreads();

        // ---- scores for 32 field rows: two 16x16 WMMA tiles, K=32 ----
#pragma unroll
        for (int frag = 0; frag < 2; ++frag) {
            V16 A;
            const unsigned short* p = &tFK[(frag * 16 + lhalf) * 48];
            A.h[0] = *reinterpret_cast<const v8bf*>(p + kA0);
            A.h[1] = *reinterpret_cast<const v8bf*>(p + 16 + kA0);
            v8f s = {};
            s = wmma_bf16(A.v, Bq.v, s);
            float e[8];
#pragma unroll
            for (int vv = 0; vv < 8; ++vv) {
                float z = s[vv] * 0.17677669529663687f;     // 1/sqrt(32)
                z = fminf(fmaxf(z, -30.f), 30.f);
                e[vv] = __expf(z);
                den += e[vv];
            }
            // field rows frag*16 + hi*8 + (0..7) are consecutive: pack pairs
            int frBase = frag * 16 + hi * 8;
#pragma unroll
            for (int vv = 0; vv < 8; vv += 2) {
                *reinterpret_cast<unsigned*>(&tW[wave][lhalf * 48 + frBase + vv]) =
                    pack_bf2(e[vv], e[vv + 1]);
            }
        }
        // same-wave LDS store->load fence (CDNA5 split counter)
        asm volatile("s_wait_dscnt 0" ::: "memory");

        // ---- exp-weight B fragment (per-wave region, no cross-wave hazard) ----
        V16 Wb;
        {
            const unsigned short* p = &tW[wave][lhalf * 48 + kB0];
            Wb.h[0] = *reinterpret_cast<const v8bf*>(p);
            Wb.h[1] = *reinterpret_cast<const v8bf*>(p + 16);
        }

        // ---- numerator: 128 value rows = 8 WMMA tiles, A from fv directly ----
#pragma unroll
        for (int vt = 0; vt < 8; ++vt) {
            const float* fp = &fvb[(size_t)(vt * 16 + lhalf) * LSEQ + f0];
            V16 A;
            A.h[0] = load8_bf(fp, kA0,      LSEQ);
            A.h[1] = load8_bf(fp, 16 + kA0, LSEQ);
            acc[vt] = wmma_bf16(A.v, Wb.v, acc[vt]);
        }
        __syncthreads();   // protect tFK before next slab overwrites it
    }

    // combine the two C-layout lane halves of each column, normalize, store
    den += __shfl_xor(den, 16);
    float inv = 1.0f / (den + 1e-16f);
    const int q = q0 + lhalf;
#pragma unroll
    for (int vt = 0; vt < 8; ++vt) {
#pragma unroll
        for (int vv = 0; vv < 8; ++vv) {
            int row = vt * 16 + hi * 8 + vv;
            zb[(size_t)row * LSEQ + q] = acc[vt][vv] * inv;
        }
    }
}

// ---------------------------------------------------------------------------
// Masked log-softmax over 16 classes: one thread per (b,l)
// ---------------------------------------------------------------------------
__global__ __launch_bounds__(256)
void logsoftmax_mask(const float* __restrict__ Yc, const int* __restrict__ classes,
                     float* __restrict__ out)
{
    int idx = blockIdx.x * 256 + threadIdx.x;
    if (idx >= 8 * 1024) return;
    int b = idx >> 10, l = idx & 1023;
    const float* yb = Yc + (size_t)b * 16 * LSEQ + l;
    int cls = classes[b];
    float v[16], mx = -1e30f;
#pragma unroll
    for (int c = 0; c < 16; ++c) {
        float t = yb[(size_t)c * LSEQ] + ((c >= cls) ? -30.0f : 0.0f);
        v[c] = t;
        mx = fmaxf(mx, t);
    }
    float s = 0.f;
#pragma unroll
    for (int c = 0; c < 16; ++c) s += __expf(v[c] - mx);
    float lse = mx + __logf(s);
    float* ob = out + (size_t)b * 16 * LSEQ + l;
#pragma unroll
    for (int c = 0; c < 16; ++c) ob[(size_t)c * LSEQ] = v[c] - lse;
}

// ---------------------------------------------------------------------------
// Host orchestration
// ---------------------------------------------------------------------------
struct CP { const float* W; const float* b; };
struct HeadP { CP fk, fv, qk; };

extern "C" void kernel_launch(void* const* d_in, const int* in_sizes, int n_in,
                              void* d_out, int out_size, void* d_ws, size_t ws_size,
                              hipStream_t stream)
{
    (void)in_sizes; (void)n_in; (void)out_size; (void)ws_size;
    const float* mem     = (const float*)d_in[0];   // [8,1,144,1024]
    const float* test    = (const float*)d_in[1];   // [8,1,128,1024]
    const int*   classes = (const int*)d_in[2];     // [8]

    int pi = 3;
    auto nextCP = [&]() { CP c{(const float*)d_in[pi], (const float*)d_in[pi + 1]}; pi += 2; return c; };
    CP emb1a = nextCP(), emb2a = nextCP(), emb1b = nextCP(), emb2b = nextCP();
    CP emb3a = nextCP(), emb3b = nextCP(), emb4a = nextCP(), emb4b = nextCP();
    CP emb5a = nextCP(), emb5b = nextCP();
    CP emb6 = nextCP(), emb7 = nextCP(), emb8 = nextCP(), emb9 = nextCP();
    HeadP heads[4][4];
    for (int blk = 0; blk < 4; ++blk)
        for (int h = 0; h < 4; ++h) {
            heads[blk][h].fk = nextCP();
            heads[blk][h].fv = nextCP();
            heads[blk][h].qk = nextCP();
        }

    // workspace (floats)
    float* ws = (float*)d_ws;
    size_t off = 0;
    auto alloc = [&](size_t n) { float* p = ws + off; off += n; return p; };
    const size_t BIG = (size_t)8 * 512 * 1024;
    float* P0 = alloc(BIG); float* P1 = alloc(BIG); float* P2 = alloc(BIG); float* P3 = alloc(BIG);
    float* FK = alloc((size_t)8 * 32 * 1024);
    float* QK = alloc((size_t)8 * 32 * 1024);
    float* FV = alloc((size_t)8 * 128 * 1024);
    float* CB = alloc((size_t)8 * 16 * 1024);

    auto conv = [&](const float* X, CP p, float* Y, int Cin, int Cout, int act,
                    const float* resid) {
        dim3 grid(1024 / 64, (Cout + 31) / 32, 8);
        conv1x1_wmma<<<grid, 128, 0, stream>>>(X, p.W, p.b, resid, Y, Cin, Cout, act);
    };
    auto mha = [&](int blk, const float* field, const float* query, float* Zout) {
        for (int h = 0; h < 4; ++h) {
            conv(field, heads[blk][h].fk, FK, 512, 32, 0, nullptr);
            conv(field, heads[blk][h].fv, FV, 512, 128, 0, nullptr);
            conv(query, heads[blk][h].qk, QK, 512, 32, 0, nullptr);
            dim3 g(1024 / 64, 8);
            attn_head_wmma<<<g, 128, 0, stream>>>(FK, FV, QK, Zout, h * 128);
        }
    };

    // x = 10*relu(conv(relu(conv(mem, emb1b)), emb2b))
    conv(mem, emb1b, P0, 144, 512, 1, nullptr);
    conv(P0, emb2b, P1, 512, 512, 2, nullptr);            // x = P1
    // y = 10*relu(conv(relu(conv(test, emb1a)), emb2a))
    conv(test, emb1a, P0, 128, 512, 1, nullptr);
    conv(P0, emb2a, P2, 512, 512, 2, nullptr);            // y = P2

    mha(0, P1, P1, P0);                                    // z = P0
    conv(P0, emb3a, P3, 512, 512, 1, nullptr);
    conv(P3, emb3b, P0, 512, 512, 0, P1);                  // x = P0

    mha(1, P0, P0, P1);
    conv(P1, emb4a, P3, 512, 512, 1, nullptr);
    conv(P3, emb4b, P1, 512, 512, 0, P0);                  // xm = P1

    mha(2, P1, P2, P0);                                    // field=xm, query=y
    conv(P0, emb5a, P3, 512, 512, 1, nullptr);
    conv(P3, emb5b, P0, 512, 512, 0, P2);                  // z = P0

    mha(3, P1, P0, P2);                                    // field=xm, query=z
    conv(P2, emb6, P3, 512, 512, 1, nullptr);
    conv(P3, emb7, P2, 512, 512, 1, nullptr);
    conv(P2, emb8, P3, 512, 512, 1, nullptr);
    conv(P3, emb9, CB, 512, 16, 0, nullptr);               // logits [8,16,1024]

    logsoftmax_mask<<<32, 256, 0, stream>>>(CB, classes, (float*)d_out);
}
